// Multihead_Attention_80178449482215
// MI455X (gfx1250) — matrix operand order repaired and checked by reading, not compiled, not run on hardware
//
#include <hip/hip_runtime.h>
#include <hip/hip_bf16.h>
#include <stdint.h>

// ---------------------------------------------------------------------------
// Multi-head attention with adaptive span, CDNA5 (gfx1250) WMMA bf16 pipeline
// B=2 S=2048 HID=1024 H=16 DH=64. All GEMMs via v_wmma_f32_16x16x32_bf16.
// Softmax row-sums are computed on the matrix unit (P x ones, E x ones)
// instead of lane shuffles; only row-max uses cross-lane ops.
// ---------------------------------------------------------------------------

typedef __attribute__((ext_vector_type(16))) __bf16 v16bf;
typedef __attribute__((ext_vector_type(8)))  float  v8f;
typedef __attribute__((ext_vector_type(4)))  unsigned int u32x4;

#define B_   2
#define S_   2048
#define HID_ 1024
#define H_   16
#define DH_  64
#define KT_  32          // k-tiles (of 32) for the 1024-deep GEMMs

union FragU { u32x4 u[2]; v16bf v; };

static __device__ __forceinline__ unsigned short f2bf(float x) {
  unsigned int u = __float_as_uint(x);
  u += 0x7FFFu + ((u >> 16) & 1u);        // round-to-nearest-even
  return (unsigned short)(u >> 16);
}

static __device__ __forceinline__ v16bf load_frag(const unsigned short* tile, int lane) {
  const u32x4* p = reinterpret_cast<const u32x4*>(tile + lane * 16);
  FragU f; f.u[0] = p[0]; f.u[1] = p[1];  // 2 x global_load_b128 per lane
  return f.v;
}

static __device__ __forceinline__ v8f wmma_bf16(v16bf a, v16bf b, v8f c) {
  return __builtin_amdgcn_wmma_f32_16x16x32_bf16(false, a, false, b, (short)0, c, false, false);
}

// Constant all-ones 32x16 bf16 B fragment (1.0bf16 = 0x3F80 in both halves).
static __device__ __forceinline__ v16bf ones_frag() {
  FragU f;
  f.u[0] = (u32x4){0x3F803F80u, 0x3F803F80u, 0x3F803F80u, 0x3F803F80u};
  f.u[1] = (u32x4){0x3F803F80u, 0x3F803F80u, 0x3F803F80u, 0x3F803F80u};
  return f.v;
}

// Inverse fragment maps (ushort index inside a 512-element tile).
// A 16x32 bf16: lanes 0-15 hold row m, K 0..7 & 16..23; lanes 16-31 K 8..15 & 24..31.
static __device__ __forceinline__ int a_frag_idx(int ml, int kl) {
  return (ml + (((kl >> 3) & 1) << 4)) * 16 + ((kl & 7) + ((kl >> 4) << 3));
}
// B 32x16 bf16: lanes 0-15 col n, K 0..7 & 16..23; lanes 16-31 K 8..15 & 24..31.
static __device__ __forceinline__ int b_frag_idx(int nl, int kl) {
  return (nl + (((kl >> 3) & 1) << 4)) * 16 + ((kl & 7) + ((kl >> 4) << 3));
}

// Max across a 16-lane half (rows of a C tile live across one half).
static __device__ __forceinline__ float halfmax(float x) {
  x = fmaxf(x, __shfl_xor(x, 1, 32));
  x = fmaxf(x, __shfl_xor(x, 2, 32));
  x = fmaxf(x, __shfl_xor(x, 4, 32));
  x = fmaxf(x, __shfl_xor(x, 8, 32));
  return x;
}

// ---------------------------------------------------------------------------
// Packing: f32 row-major -> bf16 WMMA fragment tiles.
// ---------------------------------------------------------------------------
__global__ __launch_bounds__(256) void pack_a_kernel(const float* __restrict__ src,
    unsigned short* __restrict__ dst, int kTiles, int rowStride) {
  int tid  = blockIdx.x * 256 + threadIdx.x;
  int idx  = tid & 511;
  int tile = tid >> 9;
  int kt = tile % kTiles, mTile = tile / kTiles;
  int lane = idx >> 4, e = idx & 15;
  int ml = lane & 15, hf = lane >> 4;
  int kl = e + 8 * hf + ((e >> 3) << 3);
  float v = src[(size_t)(mTile * 16 + ml) * rowStride + kt * 32 + kl];
  dst[tid] = f2bf(v);
}

__global__ __launch_bounds__(256) void pack_b_kernel(const float* __restrict__ src,
    unsigned short* __restrict__ dst, int kTiles, int rowStride) {
  int tid  = blockIdx.x * 256 + threadIdx.x;
  int idx  = tid & 511;
  int tile = tid >> 9;
  int kt = tile % kTiles, nTile = tile / kTiles;
  int lane = idx >> 4, e = idx & 15;
  int nl = lane & 15, hf = lane >> 4;
  int kl = e + 8 * hf + ((e >> 3) << 3);
  float v = src[(size_t)(kt * 32 + kl) * rowStride + nTile * 16 + nl];
  dst[tid] = f2bf(v);
}

// ---------------------------------------------------------------------------
// WMMA GEMM: M=4096, N=1024, K=1024. Wave = 64x64 tile (4x4 accums,
// 16 WMMAs per 8 fragment loads per k-step). MODE templated epilogues:
// MODE 0: -> Q packed as A-frags per (b,h)     [bh][sTile 128][dTile 2]
// MODE 1: -> K packed as B-frags (dh x key)    [bh][keyTile 128][dTile 2]
// MODE 2: -> V packed as B-frags (key x dh)    [bh][keyTile 64][dTile 4]
// MODE 3: -> f32 row-major to d_out
// ---------------------------------------------------------------------------
template <int MODE>
__global__ __launch_bounds__(256) void gemm_kernel(const unsigned short* __restrict__ A,
    const unsigned short* __restrict__ Bp,
    unsigned short* __restrict__ outPacked, float* __restrict__ outF32) {
  int lane = threadIdx.x & 31;
  int wave = threadIdx.x >> 5;
  int job  = blockIdx.x * 8 + wave;   // 1024 wave jobs
  int wn = job & 15;                  // N / 64
  int wm = job >> 4;                  // M / 64  (0..63)

  v8f acc[4][4] = {};
  #pragma unroll 1
  for (int kt = 0; kt < KT_; ++kt) {
    v16bf a[4], b[4];
    #pragma unroll
    for (int i = 0; i < 4; ++i)
      a[i] = load_frag(A + ((size_t)((wm * 4 + i) * KT_ + kt)) * 512, lane);
    #pragma unroll
    for (int j = 0; j < 4; ++j)
      b[j] = load_frag(Bp + ((size_t)((wn * 4 + j) * KT_ + kt)) * 512, lane);
    #pragma unroll
    for (int i = 0; i < 4; ++i)
      #pragma unroll
      for (int j = 0; j < 4; ++j)
        acc[i][j] = wmma_bf16(a[i], b[j], acc[i][j]);
  }

  int hf = lane >> 4, col = lane & 15;
  #pragma unroll
  for (int i = 0; i < 4; ++i)
  #pragma unroll
  for (int j = 0; j < 4; ++j)
  #pragma unroll
  for (int r = 0; r < 8; ++r) {
    int m_g = (wm * 4 + i) * 16 + r + 8 * hf;     // token index
    int n_g = (wn * 4 + j) * 16 + col;            // output column
    float v = acc[i][j][r];
    if constexpr (MODE == 3) {
      outF32[(size_t)m_g * 1024 + n_g] = v;
    } else {
      int bidx = m_g >> 11, s = m_g & 2047;
      int h = n_g >> 6, d = n_g & 63;
      int bh = bidx * 16 + h;
      unsigned short bv = f2bf(v);
      if constexpr (MODE == 0) {
        int mT = s >> 4, ml = s & 15, ktD = d >> 5, kl = d & 31;
        outPacked[((size_t)(bh * 128 + mT) * 2 + ktD) * 512 + a_frag_idx(ml, kl)] = bv;
      } else if constexpr (MODE == 1) {
        int nT = s >> 4, nl = s & 15, ktD = d >> 5, kl = d & 31;
        outPacked[((size_t)(bh * 128 + nT) * 2 + ktD) * 512 + b_frag_idx(nl, kl)] = bv;
      } else {
        int ktS = s >> 5, kl = s & 31, nT = d >> 4, nl = d & 15;
        outPacked[((size_t)(bh * 64 + ktS) * 4 + nT) * 512 + b_frag_idx(nl, kl)] = bv;
      }
    }
  }
}

// ---------------------------------------------------------------------------
// Flash-style attention. One wave per (b,h,qTile of 16 rows); 4 waves/block.
// scores = QK^T/64 + causal(-1e9); online softmax tracking Z=sum(exp) and
// L=sum(exp*span_mask); final w = exp*mask / (L + 1e-8*Z)  (exact reference
// algebra for softmax -> mask -> renormalize(+1e-8)).
// Row sums of P and E computed via WMMA against a ones fragment; output
// scattered into A-frag layout for the Wo GEMM.
// ---------------------------------------------------------------------------
__global__ __launch_bounds__(128) void attn_kernel(const unsigned short* __restrict__ QA,
    const unsigned short* __restrict__ KB, const unsigned short* __restrict__ VB,
    const float* __restrict__ span, unsigned short* __restrict__ attnA) {
  // per wave: 16x32 bf16 P tile + 16x32 bf16 E tile
  __shared__ __align__(16) unsigned short pbuf[4 * 1024];
  int lane = threadIdx.x & 31;
  int wave = threadIdx.x >> 5;
  int job = blockIdx.x * 4 + wave;     // 4096 jobs
  int qTile = job & 127;
  int bh = job >> 7;
  int h = bh & 15;
  int bb = bh >> 4;
  int hf = lane >> 4, col = lane & 15;
  unsigned short* pb = pbuf + wave * 1024;       // P tile
  unsigned short* eb = pb + 512;                 // E tile

  float spanAdd = span[h] * 2048.0f;   // span_val * attn_span
  const unsigned short* QAb = QA + (size_t)bh * 128 * 2 * 512;
  v16bf qa0 = load_frag(QAb + (qTile * 2 + 0) * 512, lane);
  v16bf qa1 = load_frag(QAb + (qTile * 2 + 1) * 512, lane);
  const v16bf onesB = ones_frag();

  float mrow[8], lrow[8], zrow[8];
  #pragma unroll
  for (int r = 0; r < 8; ++r) { mrow[r] = -3.0e38f; lrow[r] = 0.f; zrow[r] = 0.f; }
  v8f o[4] = {};

  int q0 = qTile * 16;
  int ccnt = (qTile >> 1) + 1;         // causal: key chunks of 32 up to row q0+15
  for (int c = 0; c < ccnt; ++c) {
    const unsigned short* KBc = KB + ((size_t)(bh * 128 + 2 * c)) * 2 * 512;
    const unsigned short* VBc = VB + ((size_t)(bh * 64 + c)) * 4 * 512;
    if (c + 1 < ccnt) {                // hide L2 latency of next chunk
      __builtin_prefetch(KBc + 2048 + lane * 16, 0, 3);   // -> global_prefetch_b8
      __builtin_prefetch(VBc + 2048 + lane * 16, 0, 3);
    }
    v16bf kb00 = load_frag(KBc + 0 * 512, lane);
    v16bf kb01 = load_frag(KBc + 1 * 512, lane);
    v16bf kb10 = load_frag(KBc + 2 * 512, lane);
    v16bf kb11 = load_frag(KBc + 3 * 512, lane);
    v8f s0 = {}, s1 = {};
    s0 = wmma_bf16(qa0, kb00, s0);
    s0 = wmma_bf16(qa1, kb01, s0);
    s1 = wmma_bf16(qa0, kb10, s1);
    s1 = wmma_bf16(qa1, kb11, s1);

    int n0 = c * 32 + col, n1 = n0 + 16;
    float mk0 = fminf(fmaxf(((float)n0 + spanAdd) * (1.f / 33.f), 0.f), 1.f);
    float mk1 = fminf(fmaxf(((float)n1 + spanAdd) * (1.f / 33.f), 0.f), 1.f);

    float scl[8];
    #pragma unroll
    for (int r = 0; r < 8; ++r) {
      int m_g = q0 + r + 8 * hf;
      float a  = s0[r] * (1.f / 64.f) + (n0 <= m_g ? 0.f : -1e9f);
      float bv = s1[r] * (1.f / 64.f) + (n1 <= m_g ? 0.f : -1e9f);
      float rm = halfmax(fmaxf(a, bv));
      float mnew = fmaxf(mrow[r], rm);
      float f = __expf(mrow[r] - mnew);
      mrow[r] = mnew;
      scl[r] = f;
      float e0 = __expf(a - mnew), e1 = __expf(bv - mnew);
      float p0 = e0 * mk0, p1 = e1 * mk1;
      int row = r + 8 * hf;
      // C-layout (row, col) -> LDS; re-read below in A-frag layout.
      pb[row * 32 + col]      = f2bf(p0);
      pb[row * 32 + col + 16] = f2bf(p1);
      eb[row * 32 + col]      = f2bf(e0);
      eb[row * 32 + col + 16] = f2bf(e1);
    }
    #pragma unroll
    for (int j = 0; j < 4; ++j)
      #pragma unroll
      for (int r = 0; r < 8; ++r)
        o[j][r] *= scl[r];

    asm volatile("s_wait_dscnt 0" ::: "memory");  // LDS RAW fence (same wave)

    int c0 = hf * 8;                    // A-frag K grouping per lane half
    FragU pf, ef;
    pf.u[0] = *reinterpret_cast<const u32x4*>(pb + col * 32 + c0);
    pf.u[1] = *reinterpret_cast<const u32x4*>(pb + col * 32 + c0 + 16);
    ef.u[0] = *reinterpret_cast<const u32x4*>(eb + col * 32 + c0);
    ef.u[1] = *reinterpret_cast<const u32x4*>(eb + col * 32 + c0 + 16);
    v16bf pa = pf.v;
    v16bf ea = ef.v;

    // Row sums on the matrix unit: D[m,*] = sum_k A[m,k] (replicated per col).
    v8f sP = {}, sE = {};
    sP = wmma_bf16(pa, onesB, sP);
    sE = wmma_bf16(ea, onesB, sE);
    #pragma unroll
    for (int r = 0; r < 8; ++r) {
      lrow[r] = lrow[r] * scl[r] + sP[r];
      zrow[r] = zrow[r] * scl[r] + sE[r];
    }

    #pragma unroll
    for (int j = 0; j < 4; ++j) {
      v16bf vb = load_frag(VBc + j * 512, lane);
      o[j] = wmma_bf16(pa, vb, o[j]);
    }
  }

  float invd[8];
  #pragma unroll
  for (int r = 0; r < 8; ++r) invd[r] = 1.0f / (lrow[r] + 1e-8f * zrow[r]);

  #pragma unroll
  for (int j = 0; j < 4; ++j)
  #pragma unroll
  for (int r = 0; r < 8; ++r) {
    float v = o[j][r] * invd[r];
    int token = bb * 2048 + q0 + r + 8 * hf;
    int n_g = h * 64 + j * 16 + col;
    int mT = token >> 4, ml = token & 15;
    int kT = n_g >> 5,  kl = n_g & 31;
    attnA[((size_t)(mT * 32 + kT)) * 512 + a_frag_idx(ml, kl)] = f2bf(v);
  }
}

// ---------------------------------------------------------------------------
// Host-side launch. Workspace layout (bytes), total 48 MiB:
//   qAct   @ 0         8 MiB   query as A-frags [256 mT][32 kT][512]
//   WqB    @ 8  MiB    2 MiB   weights as B-frags [64 nT][32 kT][512]
//   WkB    @ 10 MiB    2 MiB
//   WvB    @ 12 MiB    2 MiB
//   WoB    @ 14 MiB    2 MiB
//   QA     @ 16 MiB    8 MiB   per (b,h) Q A-frags
//   KBp    @ 24 MiB    8 MiB   per (b,h) K^T B-frags
//   VBp    @ 32 MiB    8 MiB   per (b,h) V B-frags
//   attnA  @ 40 MiB    8 MiB   attention output as A-frags
// ---------------------------------------------------------------------------
extern "C" void kernel_launch(void* const* d_in, const int* in_sizes, int n_in,
                              void* d_out, int out_size, void* d_ws, size_t ws_size,
                              hipStream_t stream) {
  (void)in_sizes; (void)n_in; (void)out_size; (void)ws_size;
  const float* query = (const float*)d_in[0];
  const float* Wq    = (const float*)d_in[1];
  const float* Wk    = (const float*)d_in[2];
  const float* Wv    = (const float*)d_in[3];
  const float* Wo    = (const float*)d_in[4];
  const float* spanv = (const float*)d_in[5];

  char* w = (char*)d_ws;
  const size_t MB = 1024 * 1024;
  unsigned short* qAct  = (unsigned short*)(w + 0 * MB);
  unsigned short* wqB   = (unsigned short*)(w + 8 * MB);
  unsigned short* wkB   = (unsigned short*)(w + 10 * MB);
  unsigned short* wvB   = (unsigned short*)(w + 12 * MB);
  unsigned short* woB   = (unsigned short*)(w + 14 * MB);
  unsigned short* QA    = (unsigned short*)(w + 16 * MB);
  unsigned short* KBp   = (unsigned short*)(w + 24 * MB);
  unsigned short* VBp   = (unsigned short*)(w + 32 * MB);
  unsigned short* attnA = (unsigned short*)(w + 40 * MB);

  // 1) pack activations + weights into WMMA fragment layouts (f32 -> bf16)
  pack_a_kernel<<<16384, 256, 0, stream>>>(query, qAct, 32, 1024);  // 256 mTiles
  pack_b_kernel<<<4096, 256, 0, stream>>>(Wq, wqB, 32, 1024);       // 64 nTiles
  pack_b_kernel<<<4096, 256, 0, stream>>>(Wk, wkB, 32, 1024);
  pack_b_kernel<<<4096, 256, 0, stream>>>(Wv, wvB, 32, 1024);
  pack_b_kernel<<<4096, 256, 0, stream>>>(Wo, woB, 32, 1024);

  // 2) QKV projections (WMMA), epilogues scatter into per-head packed layouts
  gemm_kernel<0><<<128, 256, 0, stream>>>(qAct, wqB, QA,  nullptr);
  gemm_kernel<1><<<128, 256, 0, stream>>>(qAct, wkB, KBp, nullptr);
  gemm_kernel<2><<<128, 256, 0, stream>>>(qAct, wvB, VBp, nullptr);

  // 3) causal + adaptive-span attention (flash-style, WMMA)
  attn_kernel<<<1024, 128, 0, stream>>>(QA, KBp, VBp, spanv, attnA);

  // 4) output projection, f32 result
  gemm_kernel<3><<<128, 256, 0, stream>>>(attnA, woB, nullptr, (float*)d_out);
}
